// Seq2Seq_13821204759182
// MI455X (gfx1250) — matrix operand order repaired and checked by reading, not compiled, not run on hardware
//
#include <hip/hip_runtime.h>
#include <hip/hip_bf16.h>
#include <cstdint>
#include <cstddef>

typedef __attribute__((ext_vector_type(16))) __bf16 v16bf;
typedef __attribute__((ext_vector_type(8)))  __bf16 v8bf;
typedef __attribute__((ext_vector_type(8)))  float  v8f;

#define BATCH  64
#define UNITS  1024
#define FEAT   512
#define KTOT   1536   /* FEAT + UNITS */
#define GATES  4096   /* 4*UNITS      */
#define TSTEPS 256
#define NT     16     /* units per workgroup -> 64 blocks per step */

// ---------------------------------------------------------------------------
// Pack [Wx ; Wh] ([F,4U] and [U,4U] row-major f32) into column-major bf16
// Wpack[n][k]: each wave's WMMA B-fragment becomes one contiguous 32B load.
// ---------------------------------------------------------------------------
__global__ __launch_bounds__(256) void pack_weights_kernel(
    const float* __restrict__ Wx, const float* __restrict__ Wh,
    __bf16* __restrict__ Wpack) {
  size_t i = (size_t)blockIdx.x * 256 + threadIdx.x;
  if (i >= (size_t)GATES * KTOT) return;
  int n = (int)(i / KTOT);
  int k = (int)(i % KTOT);
  float v = (k < FEAT) ? Wx[(size_t)k * GATES + n]
                       : Wh[(size_t)(k - FEAT) * GATES + n];
  Wpack[i] = (__bf16)v;
}

__global__ __launch_bounds__(256) void f32_to_bf16_kernel(
    const float* __restrict__ src, __bf16* __restrict__ dst, int n) {
  int i = blockIdx.x * 256 + threadIdx.x;
  if (i < n) dst[i] = (__bf16)src[i];
}

// ---------------------------------------------------------------------------
// One fused LSTM timestep:
//   z = [x_t ; h_prev] @ [Wx ; Wh] + b      (bf16 WMMA, f32 accumulate)
//   i,f,o = sigmoid ; g = tanh ; c = f*c + i*g ; h = o*tanh(c)
// Grid: UNITS/NT = 64 blocks x 256 threads (8 wave32).
// Wave w: gate = w&3, row-half = w>>2; computes a 32x16 z chunk
// (2 accumulator tiles). The 48-chunk K loop is FULLY UNROLLED with
// explicit one-chunk-ahead prefetch: SSA renaming kills buffer-rotation
// copies, the x/h pointer select constant-folds, and all loads get
// immediate offsets while retaining partial s_wait_loadcnt overlap.
// ---------------------------------------------------------------------------
__global__ __launch_bounds__(256) void lstm_step_kernel(
    const __bf16* __restrict__ xbf,    // [B, T, F] bf16
    int t,
    const __bf16* __restrict__ Wpack,  // [4096][1536] col-major bf16
    const float*  __restrict__ bias,   // [4096] f32
    const __bf16* __restrict__ h_in,   // [B, U] bf16
    __bf16* __restrict__ h_out,        // [B, U] bf16
    float* __restrict__ c,             // [B, U] f32 (updated in place)
    float* __restrict__ hs)            // [T, B, U] f32 or nullptr
{
  __shared__ float z_s[BATCH][4 * NT];   // 64 x 64 f32 = 16KB

  const int tid  = threadIdx.x;
  const int wv   = tid >> 5;        // wave 0..7
  const int lane = tid & 31;
  const int gate = wv & 3;          // 0..3 (i,f,g,o)
  const int rh   = wv >> 2;         // 0..1 (32-row half)
  const int u0   = blockIdx.x * NT;
  const int hsel = lane >> 4;       // half-wave select
  const int l16  = lane & 15;

  // B operand: this lane's weight column (absolute gate column), contiguous K
  const int ncol = gate * UNITS + u0 + l16;
  const __bf16* bcol = Wpack + (size_t)ncol * KTOT + hsel * 16;

  v8f acc[2];
  #pragma unroll
  for (int r = 0; r < 2; ++r)
    #pragma unroll
    for (int j = 0; j < 8; ++j) acc[r][j] = 0.0f;

  // A operand row base pointers (contiguous K to match B: lanes 0-15 row m,
  // K = kb+0..15; lanes 16-31: K = kb+16..31)
  const __bf16* xrow[2];
  const __bf16* hrow[2];
  #pragma unroll
  for (int r = 0; r < 2; ++r) {
    int brow = (rh * 2 + r) * 16 + l16;
    xrow[r] = xbf  + ((size_t)brow * TSTEPS + t) * FEAT + hsel * 16;
    hrow[r] = h_in + (size_t)brow * UNITS + hsel * 16;
  }

  // ---- fully-unrolled, one-chunk-ahead pipelined K loop ----
  constexpr int NCHUNK = KTOT / 32;   // 48
  v16bf bf_c = *(const v16bf*)(bcol);
  v8bf  a0c[2], a1c[2];
  #pragma unroll
  for (int r = 0; r < 2; ++r) {
    a0c[r] = *(const v8bf*)(xrow[r]);
    a1c[r] = *(const v8bf*)(xrow[r] + 8);
  }

  #pragma unroll
  for (int kc = 0; kc < NCHUNK; ++kc) {
    v16bf bf_n;
    v8bf  a0n[2], a1n[2];
    if (kc + 1 < NCHUNK) {
      const int kn = (kc + 1) * 32;   // compile-time constant after unroll
      bf_n = *(const v16bf*)(bcol + kn);
      #pragma unroll
      for (int r = 0; r < 2; ++r) {
        const __bf16* p = (kn < FEAT) ? (xrow[r] + kn) : (hrow[r] + (kn - FEAT));
        a0n[r] = *(const v8bf*)(p);
        a1n[r] = *(const v8bf*)(p + 8);
      }
    }
    #pragma unroll
    for (int r = 0; r < 2; ++r) {
      v16bf af;
      #pragma unroll
      for (int j = 0; j < 8; ++j) { af[j] = a0c[r][j]; af[8 + j] = a1c[r][j]; }
      acc[r] = __builtin_amdgcn_wmma_f32_16x16x32_bf16(
          false, af, false, bf_c, (short)0, acc[r], false, false);
    }
    if (kc + 1 < NCHUNK) {
      bf_c = bf_n;                    // renamed away by full unroll (SSA)
      #pragma unroll
      for (int r = 0; r < 2; ++r) { a0c[r] = a0n[r]; a1c[r] = a1n[r]; }
    }
  }

  // Spill z tiles to LDS (C/D layout: vgpr q, lane -> M = hsel*8+q, N = l16)
  #pragma unroll
  for (int r = 0; r < 2; ++r)
    #pragma unroll
    for (int q = 0; q < 8; ++q)
      z_s[(rh * 2 + r) * 16 + hsel * 8 + q][gate * NT + l16] = acc[r][q];
  __syncthreads();

  // Fused gate epilogue: 64 rows x 16 units = 1024 elements, 4 per thread
  for (int idx = tid; idx < BATCH * NT; idx += 256) {
    int b  = idx >> 4;          // / NT (NT == 16)
    int ul = idx & (NT - 1);
    int u  = u0 + ul;
    float zi = z_s[b][0 * NT + ul] + bias[u];
    float zf = z_s[b][1 * NT + ul] + bias[UNITS + u];
    float zg = z_s[b][2 * NT + ul] + bias[2 * UNITS + u];
    float zo = z_s[b][3 * NT + ul] + bias[3 * UNITS + u];
    float ig = 1.0f / (1.0f + __expf(-zi));
    float fg = 1.0f / (1.0f + __expf(-zf));
    float gg = tanhf(zg);
    float og = 1.0f / (1.0f + __expf(-zo));
    size_t cu = (size_t)b * UNITS + u;
    float cn  = fg * c[cu] + ig * gg;
    c[cu]     = cn;
    float hn  = og * tanhf(cn);
    h_out[cu] = (__bf16)hn;
    if (hs) hs[((size_t)t * BATCH + b) * UNITS + u] = hn;
  }
}

// ---------------------------------------------------------------------------
// Dense(1): out[t*B+b] = hs[t,b,:] . Wp + bp. One wave32 per output row.
// ---------------------------------------------------------------------------
__global__ __launch_bounds__(256) void proj_kernel(
    const float* __restrict__ hs, const float* __restrict__ Wp,
    const float* __restrict__ bp, float* __restrict__ out, int rows) {
  int g    = (blockIdx.x * 256 + threadIdx.x) >> 5;
  int lane = threadIdx.x & 31;
  if (g >= rows) return;
  const float* row = hs + (size_t)g * UNITS;
  float s = 0.0f;
  for (int u = lane; u < UNITS; u += 32) s += row[u] * Wp[u];
  #pragma unroll
  for (int o = 16; o > 0; o >>= 1) s += __shfl_down(s, o, 32);
  if (lane == 0) out[g] = s + bp[0];
}

// ---------------------------------------------------------------------------
extern "C" void kernel_launch(void* const* d_in, const int* in_sizes, int n_in,
                              void* d_out, int out_size, void* d_ws, size_t ws_size,
                              hipStream_t stream) {
  (void)in_sizes; (void)n_in; (void)out_size; (void)ws_size;

  const float* enc_x  = (const float*)d_in[0];
  const float* dec_x  = (const float*)d_in[1];
  const float* Wx_enc = (const float*)d_in[2];
  const float* Wh_enc = (const float*)d_in[3];
  const float* b_enc  = (const float*)d_in[4];
  const float* Wx_dec = (const float*)d_in[5];
  const float* Wh_dec = (const float*)d_in[6];
  const float* b_dec  = (const float*)d_in[7];
  const float* Wp     = (const float*)d_in[8];
  const float* bp     = (const float*)d_in[9];
  float* out = (float*)d_out;

  char* ws = (char*)d_ws;
  size_t off = 0;
  auto take = [&](size_t bytes) -> char* {
    char* p = ws + off;
    off += (bytes + 255) & ~(size_t)255;
    return p;
  };
  __bf16* WpackEnc = (__bf16*)take((size_t)GATES * KTOT * 2);          // 12.6 MB
  __bf16* WpackDec = (__bf16*)take((size_t)GATES * KTOT * 2);          // 12.6 MB
  __bf16* xbfEnc   = (__bf16*)take((size_t)BATCH * TSTEPS * FEAT * 2); // 16 MB
  __bf16* xbfDec   = (__bf16*)take((size_t)BATCH * TSTEPS * FEAT * 2); // 16 MB
  __bf16* hbuf0    = (__bf16*)take((size_t)BATCH * UNITS * 2);
  __bf16* hbuf1    = (__bf16*)take((size_t)BATCH * UNITS * 2);
  float*  cbuf     = (float*) take((size_t)BATCH * UNITS * 4);
  float*  decHs    = (float*) take((size_t)TSTEPS * BATCH * UNITS * 4); // 64 MB
  __bf16* hbuf[2]  = { hbuf0, hbuf1 };

  // Weight packing + input precision conversion (runs every call; deterministic)
  {
    size_t nW = (size_t)GATES * KTOT;
    int blocks = (int)((nW + 255) / 256);
    pack_weights_kernel<<<blocks, 256, 0, stream>>>(Wx_enc, Wh_enc, WpackEnc);
    pack_weights_kernel<<<blocks, 256, 0, stream>>>(Wx_dec, Wh_dec, WpackDec);
  }
  {
    int nX = BATCH * TSTEPS * FEAT;
    int blocks = (nX + 255) / 256;
    f32_to_bf16_kernel<<<blocks, 256, 0, stream>>>(enc_x, xbfEnc, nX);
    f32_to_bf16_kernel<<<blocks, 256, 0, stream>>>(dec_x, xbfDec, nX);
  }
  hipMemsetAsync(hbuf[0], 0, (size_t)BATCH * UNITS * 2, stream);  // h0 = 0
  hipMemsetAsync(cbuf,    0, (size_t)BATCH * UNITS * 4, stream);  // c0 = 0

  // Encoder: 256 sequential fused steps (h ping-pongs; ends in hbuf[0])
  for (int t = 0; t < TSTEPS; ++t)
    lstm_step_kernel<<<UNITS / NT, 256, 0, stream>>>(
        xbfEnc, t, WpackEnc, b_enc, hbuf[t & 1], hbuf[(t + 1) & 1], cbuf,
        (float*)nullptr);

  // Decoder: seeded by (hT, cT); also records f32 h into decHs
  for (int t = 0; t < TSTEPS; ++t)
    lstm_step_kernel<<<UNITS / NT, 256, 0, stream>>>(
        xbfDec, t, WpackDec, b_dec, hbuf[t & 1], hbuf[(t + 1) & 1], cbuf, decHs);

  // Final Dense(1) projection: [T,B,1]
  {
    int rows = TSTEPS * BATCH;                 // 16384
    int blocks = (rows * 32 + 255) / 256;      // one wave per row
    proj_kernel<<<blocks, 256, 0, stream>>>(decHs, Wp, bp, out, rows);
  }
}
